// Model_67516885893194
// MI455X (gfx1250) — compile-verified, matrix operand order checked by reading, not run on hardware
//
#include <hip/hip_runtime.h>
#include <hip/hip_bf16.h>
#include <cmath>
#include <cstddef>

// ---- problem constants (match reference) ----
#define B_   4096
#define T_   256
#define D_   36
#define H_   64
#define NI_  192     // 3*H
#define KX_  128     // K pad for x (108 -> 128)
#define ROWS 64      // batch rows per block
#define NTHREADS 128 // 4 waves * 32 lanes; each wave owns 16 rows
#define EPSF 1e-5f

typedef __attribute__((ext_vector_type(16))) __bf16 bf16x16;
typedef __attribute__((ext_vector_type(8)))  __bf16 bf16x8;
typedef __attribute__((ext_vector_type(8)))  float  f32x8;
typedef __attribute__((ext_vector_type(4)))  unsigned int u32x4;
typedef __attribute__((ext_vector_type(8)))  int    i32x8;
typedef __attribute__((ext_vector_type(4)))  int    i32x4;

// Dynamic LDS layout (~204 KB; WGP has 320 KB)
struct __align__(16) Smem {
  __bf16 Wih [NI_ * KX_];     // [192][128]  (K=108 zero-padded)
  __bf16 Whh [NI_ * H_];      // [192][64]
  __bf16 Wdec[48 * 64];       // [48][64]    (36x36 padded)
  __bf16 Wcomb[48 * 96];      // [48][96]    (36x72 padded)
  __bf16 Wfeat[48 * 64];      // [48][64]    (36x36, zero diag, padded)
  __bf16 actA[ROWS * KX_];    // activation panel, reused per GEMM
  __bf16 hbf [ROWS * H_];     // hidden state, bf16 copy for WMMA
  float  hf32[ROWS * H_];     // hidden state, f32 master
  float  inb[2][3][ROWS * D_];// TDM double buffer: [buf][values|masks|deltas]
  float  vcS [ROWS * D_];     // first value_c
  float  vx  [ROWS];          // value_x per row
  float  logits[ROWS];
  float  bih[NI_], bhh[NI_];
  float  bdec[D_], bcomb[D_], bfeat[D_];
  float  histW[H_], outW[H_];
  float  red[NTHREADS];
  float  hist_b, out_b;
};

__device__ __forceinline__ bf16x16 load_frag(const __bf16* base, int stride,
                                             int row0, int k0, int lane) {
  // 16-bit WMMA operand layout (A and B identical with row = M or N):
  // lanes 0-15: k in {k0..k0+7} U {k0+16..k0+23}; lanes 16-31: +8
  const __bf16* p = base + (row0 + (lane & 15)) * stride + k0 + ((lane >> 4) << 3);
  bf16x8 lo = *(const bf16x8*)p;
  bf16x8 hi = *(const bf16x8*)(p + 16);
  return __builtin_shufflevector(lo, hi, 0,1,2,3,4,5,6,7,8,9,10,11,12,13,14,15);
}

__device__ __forceinline__ f32x8 wmma_bf16(bf16x16 a, bf16x16 b, f32x8 c) {
  return __builtin_amdgcn_wmma_f32_16x16x32_bf16(false, a, false, b,
                                                 (short)0, c, false, false);
}

__device__ __forceinline__ void load_w(__bf16* dst, const float* src,
                                       int N, int K, int Npad, int Kpad,
                                       int tid, bool zero_diag) {
  for (int i = tid; i < Npad * Kpad; i += NTHREADS) {
    int n = i / Kpad, k = i - n * Kpad;
    float v = (n < N && k < K) ? src[n * K + k] : 0.f;
    if (zero_diag && n == k) v = 0.f;
    dst[i] = (__bf16)v;
  }
}

__device__ __forceinline__ float sigm(float x) { return 1.f / (1.f + __expf(-x)); }

// Tensor Data Mover: 2D tile load, 64 rows x 36 f32, row stride T*D elements.
// D# per CDNA5 ISA ch.8: group0 {count, lds_addr, global_addr, type=2},
// group1 {data_size=4B, tensor_dim0/1, tile_dim0/1, dim0 stride}; groups 2/3 zero
// (2-D tile, tile_dim2=0). 6-arg builtin form (clang-23 / therock-10.0 headers).
__device__ __forceinline__ void tdm_load_2d(unsigned int lds_off, const float* gptr) {
  unsigned long long ga = (unsigned long long)(size_t)gptr;
  u32x4 g0;
  g0[0] = 1u;                                   // count=1 (valid descriptor)
  g0[1] = lds_off;                              // LDS byte address of tile
  g0[2] = (unsigned int)(ga & 0xffffffffull);   // global_addr[31:0]
  g0[3] = (unsigned int)((ga >> 32) & 0x1ffffffull) | 0x80000000u; // [56:32] | type=2
  i32x8 g1;
  g1[0] = (int)(2u << 16);                      // data_size = 2 (4 bytes)
  g1[1] = (int)((unsigned)D_ << 16);            // tensor_dim0[15:0] = 36
  g1[2] = (int)((unsigned)ROWS << 16);          // tensor_dim1[15:0] = 64
  g1[3] = (int)((unsigned)D_ << 16);            // tile_dim0 = 36
  g1[4] = (int)(unsigned)ROWS;                  // tile_dim1 = 64, tile_dim2 = 0
  g1[5] = (int)(T_ * D_);                       // tensor_dim0_stride[31:0] = 9216
  g1[6] = 0;                                    // stride0[47:32]=0, stride1 lo=0
  g1[7] = 0;
  i32x4 z4 = {0, 0, 0, 0};
  i32x8 z8 = {0, 0, 0, 0, 0, 0, 0, 0};
  __builtin_amdgcn_tensor_load_to_lds(g0, g1, z4, z4, z8, 0);
}

// ---------------- prep: per-t mask sums (+EPS), sum(w)+EPS, zero loss ----
__global__ __launch_bounds__(256)
void prep_kernel(const float* __restrict__ masks, const float* __restrict__ is_train,
                 float* __restrict__ ws, float* __restrict__ out0) {
  __shared__ float red[256];
  const int t = blockIdx.x, tid = threadIdx.x;
  float s = 0.f;
  if (t < T_) {
    for (int n = tid; n < B_ * D_; n += 256) {
      int b = n / D_, d = n - b * D_;
      s += masks[(size_t)b * (T_ * D_) + (size_t)t * D_ + d];
    }
  } else {
    for (int n = tid; n < B_; n += 256) s += is_train[n];
    if (tid == 0) *out0 = 0.f;
  }
  red[tid] = s; __syncthreads();
  for (int off = 128; off > 0; off >>= 1) {
    if (tid < off) red[tid] += red[tid + off];
    __syncthreads();
  }
  if (tid == 0) ws[t] = red[0] + EPSF;
}

// ---------------- main persistent RNN kernel ----------------------------
__global__ __launch_bounds__(NTHREADS)
void brits_kernel(const float* __restrict__ values, const float* __restrict__ masks,
                  const float* __restrict__ deltas, const float* __restrict__ label,
                  const float* __restrict__ is_train,
                  const float* __restrict__ W_ih, const float* __restrict__ W_hh,
                  const float* __restrict__ b_ih, const float* __restrict__ b_hh,
                  const float* __restrict__ hist_W, const float* __restrict__ hist_bp,
                  const float* __restrict__ feat_W, const float* __restrict__ feat_b,
                  const float* __restrict__ decay_W, const float* __restrict__ decay_b,
                  const float* __restrict__ comb_W, const float* __restrict__ comb_b,
                  const float* __restrict__ out_W, const float* __restrict__ out_bp,
                  const float* __restrict__ msum,   // ws: [0..T-1] mask sums, [T] sum(w)
                  float* __restrict__ out) {        // [0] loss | impute | pred
  extern __shared__ char smraw[];
  Smem& sm = *(Smem*)smraw;
  const int tid  = threadIdx.x;
  const int lane = tid & 31;
  const int wave = tid >> 5;
  const int rowBase = wave * 16;          // this wave's 16-row M tile
  const int bBase = blockIdx.x * ROWS;    // global batch base
  const size_t gBase = (size_t)bBase * (T_ * D_);

  // ---- init: weights -> bf16 LDS (zero-padded), state -> 0 ----
  load_w(sm.Wih,  W_ih,  NI_, 3 * D_, NI_, KX_, tid, false);
  load_w(sm.Whh,  W_hh,  NI_, H_,     NI_, H_,  tid, false);
  load_w(sm.Wdec, decay_W, D_, D_, 48, 64, tid, false);
  load_w(sm.Wcomb, comb_W, D_, 2 * D_, 48, 96, tid, false);
  load_w(sm.Wfeat, feat_W, D_, D_, 48, 64, tid, true);   // zero diagonal
  for (int i = tid; i < ROWS * KX_; i += NTHREADS) sm.actA[i] = (__bf16)0.f;
  for (int i = tid; i < ROWS * H_;  i += NTHREADS) { sm.hbf[i] = (__bf16)0.f; sm.hf32[i] = 0.f; }
  for (int i = tid; i < NI_; i += NTHREADS) { sm.bih[i] = b_ih[i]; sm.bhh[i] = b_hh[i]; }
  for (int i = tid; i < D_;  i += NTHREADS) { sm.bdec[i] = decay_b[i]; sm.bcomb[i] = comb_b[i]; sm.bfeat[i] = feat_b[i]; }
  for (int i = tid; i < H_;  i += NTHREADS) { sm.histW[i] = hist_W[i]; sm.outW[i] = out_W[i]; }
  if (tid == 0) { sm.hist_b = hist_bp[0]; sm.out_b = out_bp[0]; }

  const unsigned inOff = (unsigned)offsetof(Smem, inb);
  const unsigned bufBytes = 3u * ROWS * D_ * 4u;   // one buffer (3 arrays)
  const unsigned arrBytes = (unsigned)(ROWS * D_ * 4);
  // TDM prefetch for t = 0 into buffer 0 (wave 0 only; TENSORcnt is per-wave)
  if (wave == 0) {
    tdm_load_2d(inOff + 0 * arrBytes, values + gBase);
    tdm_load_2d(inOff + 1 * arrBytes, masks  + gBase);
    tdm_load_2d(inOff + 2 * arrBytes, deltas + gBase);
  }
  __syncthreads();

  float hsum[4][8];
  #pragma unroll
  for (int a = 0; a < 4; ++a)
    #pragma unroll
    for (int b = 0; b < 8; ++b) hsum[a][b] = 0.f;
  float imputeAcc = 0.f, classAcc = 0.f;

  for (int t = 0; t < T_; ++t) {
    // (0) kick TDM for t+1 into the other buffer, then wait for t's tiles.
    if (wave == 0) {
      if (t + 1 < T_) {
        const unsigned off = inOff + ((t + 1) & 1) * bufBytes;
        const size_t g = gBase + (size_t)(t + 1) * D_;
        tdm_load_2d(off + 0 * arrBytes, values + g);
        tdm_load_2d(off + 1 * arrBytes, masks  + g);
        tdm_load_2d(off + 2 * arrBytes, deltas + g);
        __builtin_amdgcn_s_wait_tensorcnt(3);   // t's 3 tiles done; t+1's in flight
      } else {
        __builtin_amdgcn_s_wait_tensorcnt(0);
      }
    }
    __syncthreads();                            // publish TDM-written LDS to all waves
    const float* vb = sm.inb[t & 1][0];
    const float* mb = sm.inb[t & 1][1];
    const float* db = sm.inb[t & 1][2];
    const float inv_m = 1.f / msum[t];

    // (1) value_x = h @ hist_W.T + hist_b  (one row per lane 0..15)
    if (lane < 16) {
      const int row = rowBase + lane;
      float s = sm.hist_b;
      #pragma unroll
      for (int k = 0; k < H_; ++k) s += sm.hf32[row * H_ + k] * sm.histW[k];
      sm.vx[row] = s;
    }
    __builtin_amdgcn_wave_barrier();

    // (2) stage inputs: delta->actA[0..35], mask->actA[36..71]; loss term 1
    for (int i = lane; i < 16 * D_; i += 32) {
      const int r = i / D_, d = i - r * D_;
      const int row = rowBase + r;
      const float v = vb[row * D_ + d], m = mb[row * D_ + d], dl = db[row * D_ + d];
      sm.actA[row * KX_ + d] = (__bf16)dl;
      sm.actA[row * KX_ + D_ + d] = (__bf16)m;
      const float vxv = sm.vx[row];
      imputeAcc += fabsf(vxv - v) * m * inv_m;
      sm.vcS[row * D_ + d] = m * v + (1.f - m) * vxv;
    }
    __builtin_amdgcn_wave_barrier();

    // (3) gamma = exp(-relu(delta @ Wdec.T + b)); write to actA[0..35] & [72..107]
    f32x8 accD[3] = {};
    #pragma unroll
    for (int ks = 0; ks < 2; ++ks) {
      const bf16x16 a = load_frag(sm.actA, KX_, rowBase, ks * 32, lane);
      #pragma unroll
      for (int nt = 0; nt < 3; ++nt)
        accD[nt] = wmma_bf16(a, load_frag(sm.Wdec, 64, nt * 16, ks * 32, lane), accD[nt]);
    }
    #pragma unroll
    for (int nt = 0; nt < 3; ++nt) {
      const int col = nt * 16 + (lane & 15);
      if (col < D_) {
        #pragma unroll
        for (int v = 0; v < 8; ++v) {
          const int row = rowBase + v + ((lane >> 4) << 3);
          const float gam = __expf(-fmaxf(accD[nt][v] + sm.bdec[col], 0.f));
          const __bf16 gb = (__bf16)gam;
          sm.actA[row * KX_ + col] = gb;        // comb input (gamma)
          sm.actA[row * KX_ + 72 + col] = gb;   // x input (gamma)
        }
      }
    }
    __builtin_amdgcn_wave_barrier();

    // (4) beta = sigmoid([gamma|mask] @ Wcomb.T + b)  (K=96 padded)
    f32x8 accC[3] = {};
    #pragma unroll
    for (int ks = 0; ks < 3; ++ks) {
      const bf16x16 a = load_frag(sm.actA, KX_, rowBase, ks * 32, lane);
      #pragma unroll
      for (int nt = 0; nt < 3; ++nt)
        accC[nt] = wmma_bf16(a, load_frag(sm.Wcomb, 96, nt * 16, ks * 32, lane), accC[nt]);
    }
    float beta[3][8];
    #pragma unroll
    for (int nt = 0; nt < 3; ++nt) {
      const int col = nt * 16 + (lane & 15);
      #pragma unroll
      for (int v = 0; v < 8; ++v)
        beta[nt][v] = (col < D_) ? sigm(accC[nt][v] + sm.bcomb[col]) : 0.f;
    }

    // stage value_c into actA[0..35] for feat GEMM
    for (int i = lane; i < 16 * D_; i += 32) {
      const int r = i / D_, d = i - r * D_;
      const int row = rowBase + r;
      sm.actA[row * KX_ + d] = (__bf16)sm.vcS[row * D_ + d];
    }
    __builtin_amdgcn_wave_barrier();

    // (5) value_z = value_c @ Wfeat.T + b;  loss 2/3; value_c2 -> out + actA
    f32x8 accF[3] = {};
    #pragma unroll
    for (int ks = 0; ks < 2; ++ks) {
      const bf16x16 a = load_frag(sm.actA, KX_, rowBase, ks * 32, lane);
      #pragma unroll
      for (int nt = 0; nt < 3; ++nt)
        accF[nt] = wmma_bf16(a, load_frag(sm.Wfeat, 64, nt * 16, ks * 32, lane), accF[nt]);
    }
    #pragma unroll
    for (int nt = 0; nt < 3; ++nt) {
      const int col = nt * 16 + (lane & 15);
      if (col < D_) {
        #pragma unroll
        for (int v = 0; v < 8; ++v) {
          const int row = rowBase + v + ((lane >> 4) << 3);
          const float vz  = accF[nt][v] + sm.bfeat[col];
          const float vL  = vb[row * D_ + col];
          const float mk  = mb[row * D_ + col];
          const float bt  = beta[nt][v];
          const float vh  = vz * bt + sm.vx[row] * (1.f - bt);
          imputeAcc += (fabsf(vz - vL) + fabsf(vh - vL)) * mk * inv_m;
          const float vc2 = mk * vL + (1.f - mk) * vh;
          out[1 + (size_t)(bBase + row) * (T_ * D_) + (size_t)t * D_ + col] = vc2;
          sm.actA[row * KX_ + col] = (__bf16)vc2;  // x input (value_c)
        }
      }
    }
    __builtin_amdgcn_wave_barrier();

    // (6) GRU: gi = x @ Wih.T (K=128), gh = h @ Whh.T (K=64).
    // r/z parts of gh folded into accG[0..7]; n part kept separate in accN.
    f32x8 accG[12] = {};
    f32x8 accN[4]  = {};
    #pragma unroll
    for (int ks = 0; ks < 4; ++ks) {
      const bf16x16 a = load_frag(sm.actA, KX_, rowBase, ks * 32, lane);
      #pragma unroll
      for (int nt = 0; nt < 12; ++nt)
        accG[nt] = wmma_bf16(a, load_frag(sm.Wih, KX_, nt * 16, ks * 32, lane), accG[nt]);
    }
    #pragma unroll
    for (int ks = 0; ks < 2; ++ks) {
      const bf16x16 a = load_frag(sm.hbf, H_, rowBase, ks * 32, lane);
      #pragma unroll
      for (int nt = 0; nt < 8; ++nt)
        accG[nt] = wmma_bf16(a, load_frag(sm.Whh, H_, nt * 16, ks * 32, lane), accG[nt]);
      #pragma unroll
      for (int nt = 8; nt < 12; ++nt)
        accN[nt - 8] = wmma_bf16(a, load_frag(sm.Whh, H_, nt * 16, ks * 32, lane), accN[nt - 8]);
    }
    #pragma unroll
    for (int nt = 0; nt < 4; ++nt) {
      const int col = nt * 16 + (lane & 15);
      #pragma unroll
      for (int v = 0; v < 8; ++v) {
        const int row = rowBase + v + ((lane >> 4) << 3);
        const float r = sigm(accG[nt][v]     + sm.bih[col]       + sm.bhh[col]);
        const float z = sigm(accG[nt + 4][v] + sm.bih[col + 64]  + sm.bhh[col + 64]);
        const float n = tanhf(accG[nt + 8][v] + sm.bih[col + 128]
                              + r * (accN[nt][v] + sm.bhh[col + 128]));
        const float hN = (1.f - z) * n + z * sm.hf32[row * H_ + col];
        sm.hf32[row * H_ + col] = hN;
        sm.hbf[row * H_ + col]  = (__bf16)hN;
        hsum[nt][v] += hN;
      }
    }
    // protect inb[t&1] from being overwritten by the TDM issued at step t+1
    __syncthreads();
  } // t loop

  // ---- finalize: logits, pred, BCE, loss reduction ----
  for (int i = tid; i < ROWS; i += NTHREADS) sm.logits[i] = 0.f;
  __syncthreads();
  const float invT = 1.f / (float)T_;
  #pragma unroll
  for (int nt = 0; nt < 4; ++nt) {
    const int col = nt * 16 + (lane & 15);
    #pragma unroll
    for (int v = 0; v < 8; ++v) {
      const int row = rowBase + v + ((lane >> 4) << 3);
      atomicAdd(&sm.logits[row], hsum[nt][v] * invT * sm.outW[col]);
    }
  }
  __syncthreads();
  const float inv_w = 1.f / msum[T_];    // sum(is_train) + EPS
  if (tid < ROWS) {
    const int b = bBase + tid;
    const float l = sm.logits[tid] + sm.out_b;
    out[1 + (size_t)B_ * T_ * D_ + b] = sigm(l);                 // pred
    const float y = label[b], w = is_train[b];
    const float bce = fmaxf(l, 0.f) - l * y + log1pf(__expf(-fabsf(l)));
    classAcc += w * bce * inv_w;
  }
  sm.red[tid] = imputeAcc * invT + classAcc;
  __syncthreads();
  if (tid == 0) {
    float s = 0.f;
    for (int i = 0; i < NTHREADS; ++i) s += sm.red[i];
    atomicAdd(out, s);
  }
}

// ---------------- host launcher ------------------------------------------
extern "C" void kernel_launch(void* const* d_in, const int* in_sizes, int n_in,
                              void* d_out, int out_size, void* d_ws, size_t ws_size,
                              hipStream_t stream) {
  const float* values   = (const float*)d_in[0];
  const float* masks    = (const float*)d_in[1];
  const float* deltas   = (const float*)d_in[2];
  const float* label    = (const float*)d_in[3];
  const float* is_train = (const float*)d_in[4];
  const float* W_ih     = (const float*)d_in[5];
  const float* W_hh     = (const float*)d_in[6];
  const float* b_ih     = (const float*)d_in[7];
  const float* b_hh     = (const float*)d_in[8];
  const float* hist_W   = (const float*)d_in[9];
  const float* hist_b   = (const float*)d_in[10];
  const float* feat_W   = (const float*)d_in[11];
  const float* feat_b   = (const float*)d_in[12];
  const float* decay_W  = (const float*)d_in[13];
  const float* decay_b  = (const float*)d_in[14];
  const float* comb_W   = (const float*)d_in[15];
  const float* comb_b   = (const float*)d_in[16];
  const float* out_W    = (const float*)d_in[17];
  const float* out_b    = (const float*)d_in[18];
  float* ws  = (float*)d_ws;
  float* out = (float*)d_out;

  (void)in_sizes; (void)n_in; (void)out_size; (void)ws_size;

  const size_t smbytes = sizeof(Smem);
  // idempotent + deterministic, not a stream op; safe under graph capture
  (void)hipFuncSetAttribute((const void*)brits_kernel,
                            hipFuncAttributeMaxDynamicSharedMemorySize,
                            (int)smbytes);

  prep_kernel<<<T_ + 1, 256, 0, stream>>>(masks, is_train, ws, out);
  brits_kernel<<<B_ / ROWS, NTHREADS, smbytes, stream>>>(
      values, masks, deltas, label, is_train,
      W_ih, W_hh, b_ih, b_hh, hist_W, hist_b,
      feat_W, feat_b, decay_W, decay_b,
      comb_W, comb_b, out_W, out_b, ws, out);
}
